// DPConformer_46651934769890
// MI455X (gfx1250) — compile-verified, matrix operand order checked by reading, not run on hardware
//
#include <hip/hip_runtime.h>
#include <math.h>

typedef _Float16 f16;
typedef __attribute__((ext_vector_type(16))) _Float16 v16h;
typedef __attribute__((ext_vector_type(8)))  float    v8f;

#define NHEAD 4
#define TREAL 250
#define TP    256      // padded T
#define CDIM  256
#define DK    64
#define FDIM  64
#define NSEQ  256      // B*F
#define MP    (NSEQ*TP)   // 65536 padded rows
#define BH    (NSEQ*NHEAD)
#define RP    512      // padded rel-position range
#define KCONV 33

// ---------------------------------------------------------------------------
// Generic WMMA GEMM:  C = epi(A(MxK,f16) * B^T(NxK,f16) + bias) [+ residual]
// Block tile 128x64 (8 waves, 32x32 per wave = 4 x v_wmma_f32_16x16x32_f16
// per 32-K step), double-buffered LDS, one barrier per K-step.
// ---------------------------------------------------------------------------
struct GemmP {
    const f16* A; const f16* B; const float* bias;
    float* Cf; f16* Ch; const float* R;
    int M, N, K, lda, ldb, ldc;
    long long sAo, sAi, sBo, sBi, sCo, sCi;   // batch strides: z -> (zo=z/innerB, zi=z%innerB)
    int innerB;
    float resScale;
    int epi;                    // 0 none, 1 relu, 2 attention-scores
    const f16* G;               // rel-pos bias matrix (bh, TP, RP) f16
    const int* mask;            // (NSEQ, TREAL, TREAL) int32
    long long sG;
    float scaleS;
};

__device__ __forceinline__ v16h ld_frag(const f16* S, int row, int h8) {
    union { v16h v; float4 q[2]; } u;
    u.q[0] = *(const float4*)(S + row * 32 + h8);
    u.q[1] = *(const float4*)(S + row * 32 + h8 + 16);
    return u.v;
}

__global__ __launch_bounds__(256)
void gemm_wmma_kernel(GemmP p) {
    const int z  = blockIdx.z;
    const int zo = z / p.innerB;
    const int zi = z - zo * p.innerB;
    const f16* A = p.A + zo * p.sAo + zi * p.sAi;
    const f16* B = p.B + zo * p.sBo + zi * p.sBi;
    const long long cOff = zo * p.sCo + zi * p.sCi;

    const int m0 = blockIdx.y * 128;
    const int n0 = blockIdx.x * 64;

    // double-buffered tiles: A 128x32 halves, B 64x32 halves
    __shared__ __align__(16) f16 As[2][128 * 32];
    __shared__ __align__(16) f16 Bs[2][64 * 32];

    const int tid  = threadIdx.x;
    const int lrow = tid >> 2;            // 0..63
    const int lchk = (tid & 3) * 8;       // 0,8,16,24 halves
    const int wave = tid >> 5;            // 0..7
    const int lane = tid & 31;
    const int mw   = wave & 3;            // m sub-block (x32)
    const int nw   = wave >> 2;           // n sub-block (x32)
    const int lr   = lane & 15;
    const int hf   = lane >> 4;
    const int h8   = hf * 8;

    v8f acc[2][2];
#pragma unroll
    for (int a = 0; a < 2; ++a)
#pragma unroll
        for (int b = 0; b < 2; ++b)
#pragma unroll
            for (int j = 0; j < 8; ++j) acc[a][b][j] = 0.0f;

    const f16* Arow0 = A + (long long)(m0 + lrow) * p.lda + lchk;          // rows 0..63
    const f16* Arow1 = A + (long long)(m0 + 64 + lrow) * p.lda + lchk;     // rows 64..127
    const f16* Brow  = B + (long long)(n0 + lrow) * p.ldb + lchk;

    // prologue: fetch + store tile 0
    float4 ra0 = *(const float4*)(Arow0);
    float4 ra1 = *(const float4*)(Arow1);
    float4 rb  = *(const float4*)(Brow);
    *(float4*)(As[0] + lrow * 32 + lchk)        = ra0;
    *(float4*)(As[0] + (64 + lrow) * 32 + lchk) = ra1;
    *(float4*)(Bs[0] + lrow * 32 + lchk)        = rb;
    __syncthreads();

    int cur = 0;
    for (int k0 = 0; k0 < p.K; k0 += 32) {
        const bool more = (k0 + 32) < p.K;
        if (more) {   // fetch next tile while computing current
            ra0 = *(const float4*)(Arow0 + k0 + 32);
            ra1 = *(const float4*)(Arow1 + k0 + 32);
            rb  = *(const float4*)(Brow  + k0 + 32);
        }

        v16h a0 = ld_frag(As[cur], 32 * mw + lr, h8);
        v16h a1 = ld_frag(As[cur], 32 * mw + 16 + lr, h8);
        v16h b0 = ld_frag(Bs[cur], 32 * nw + lr, h8);
        v16h b1 = ld_frag(Bs[cur], 32 * nw + 16 + lr, h8);
        acc[0][0] = __builtin_amdgcn_wmma_f32_16x16x32_f16(false, a0, false, b0,
                                                           (short)0, acc[0][0], false, false);
        acc[0][1] = __builtin_amdgcn_wmma_f32_16x16x32_f16(false, a0, false, b1,
                                                           (short)0, acc[0][1], false, false);
        acc[1][0] = __builtin_amdgcn_wmma_f32_16x16x32_f16(false, a1, false, b0,
                                                           (short)0, acc[1][0], false, false);
        acc[1][1] = __builtin_amdgcn_wmma_f32_16x16x32_f16(false, a1, false, b1,
                                                           (short)0, acc[1][1], false, false);

        if (more) {   // store next tile into the other buffer
            const int nxt = cur ^ 1;
            *(float4*)(As[nxt] + lrow * 32 + lchk)        = ra0;
            *(float4*)(As[nxt] + (64 + lrow) * 32 + lchk) = ra1;
            *(float4*)(Bs[nxt] + lrow * 32 + lchk)        = rb;
        }
        __syncthreads();
        cur ^= 1;
    }

    // epilogue
#pragma unroll
    for (int ms = 0; ms < 2; ++ms) {
#pragma unroll
        for (int ns = 0; ns < 2; ++ns) {
            const int ng = n0 + 32 * nw + ns * 16 + lr;
#pragma unroll
            for (int j = 0; j < 8; ++j) {
                const int mg = m0 + 32 * mw + ms * 16 + 8 * hf + j;
                float v = acc[ms][ns][j];
                if (p.bias) v += p.bias[ng];
                if (p.epi == 1) {
                    v = fmaxf(v, 0.0f);
                } else if (p.epi == 2) {
                    // attention scores: v = (qk + G[bh,t,t-s+249]) * scale, masked
                    int rel = mg - ng + 249;
                    rel = rel < 0 ? 0 : (rel > RP - 1 ? RP - 1 : rel);
                    float g = (float)p.G[(long long)z * p.sG + (long long)mg * RP + rel];
                    v = (v + g) * p.scaleS;
                    bool ok = (mg < TREAL) && (ng < TREAL);
                    if (ok) ok = (p.mask[(long long)zo * (TREAL * TREAL) +
                                         (long long)mg * TREAL + ng] != 0);
                    if (!ok) v = -3.0e38f;
                }
                const long long off = cOff + (long long)mg * p.ldc + ng;
                if (p.R)  v = p.R[off] + p.resScale * v;
                if (p.Cf) p.Cf[off] = v;
                if (p.Ch) p.Ch[off] = (f16)v;
            }
        }
    }
}

// ---------------------------------------------------------------------------
// LayerNorm over C=256, one wave per row, f32 in -> f16 out
// ---------------------------------------------------------------------------
__global__ __launch_bounds__(256)
void ln_kernel(const float* __restrict__ X, const float* __restrict__ g,
               const float* __restrict__ b, f16* __restrict__ O, int rows) {
    const int wave = threadIdx.x >> 5, lane = threadIdx.x & 31;
    const long long r = (long long)blockIdx.x * 8 + wave;
    if (r >= rows) return;
    const float* x = X + r * CDIM + lane * 8;
    float4 a0 = *(const float4*)x;
    float4 a1 = *(const float4*)(x + 4);
    float v[8] = {a0.x, a0.y, a0.z, a0.w, a1.x, a1.y, a1.z, a1.w};
    float s = 0.f, q = 0.f;
#pragma unroll
    for (int i = 0; i < 8; ++i) { s += v[i]; q += v[i] * v[i]; }
    for (int m = 16; m >= 1; m >>= 1) { s += __shfl_xor(s, m, 32); q += __shfl_xor(q, m, 32); }
    const float mean = s * (1.f / CDIM);
    const float var  = q * (1.f / CDIM) - mean * mean;
    const float rs   = rsqrtf(var + 1e-5f);
    union { f16 h[8]; float4 f; } o;
#pragma unroll
    for (int i = 0; i < 8; ++i) {
        const int c = lane * 8 + i;
        o.h[i] = (f16)((v[i] - mean) * rs * g[c] + b[c]);
    }
    *(float4*)(O + r * CDIM + lane * 8) = o.f;
}

// Final LayerNorm + scatter to (B,T,C,F) f32 output. One wave per real row.
__global__ __launch_bounds__(256)
void ln_final_kernel(const float* __restrict__ X, const float* __restrict__ g,
                     const float* __restrict__ b, float* __restrict__ out, int rows) {
    const int wave = threadIdx.x >> 5, lane = threadIdx.x & 31;
    const long long r = (long long)blockIdx.x * 8 + wave;
    if (r >= rows) return;
    const int nn = (int)(r / TREAL);     // sequence index (b*F + f)
    const int t  = (int)(r % TREAL);
    const int bb = nn >> 6;
    const int ff = nn & 63;
    const float* x = X + ((long long)nn * TP + t) * CDIM + lane * 8;
    float4 a0 = *(const float4*)x;
    float4 a1 = *(const float4*)(x + 4);
    float v[8] = {a0.x, a0.y, a0.z, a0.w, a1.x, a1.y, a1.z, a1.w};
    float s = 0.f, q = 0.f;
#pragma unroll
    for (int i = 0; i < 8; ++i) { s += v[i]; q += v[i] * v[i]; }
    for (int m = 16; m >= 1; m >>= 1) { s += __shfl_xor(s, m, 32); q += __shfl_xor(q, m, 32); }
    const float mean = s * (1.f / CDIM);
    const float var  = q * (1.f / CDIM) - mean * mean;
    const float rs   = rsqrtf(var + 1e-5f);
#pragma unroll
    for (int i = 0; i < 8; ++i) {
        const int c = lane * 8 + i;
        const float y = (v[i] - mean) * rs * g[c] + b[c];
        out[(((long long)bb * TREAL + t) * CDIM + c) * FDIM + ff] = y;
    }
}

// (B,T,C,F) f32 -> padded (NSEQ, TP, C) f32, zero pad rows
__global__ __launch_bounds__(256)
void pad_input_kernel(const float* __restrict__ x, float* __restrict__ y, int n) {
    const int idx = blockIdx.x * 256 + threadIdx.x;
    if (idx >= n) return;
    const int c = idx & 255;
    const int row = idx >> 8;
    const int t = row & 255;
    const int nn = row >> 8;
    const int bb = nn >> 6, ff = nn & 63;
    float v = 0.f;
    if (t < TREAL)
        v = x[(((long long)bb * TREAL + t) * CDIM + c) * FDIM + ff];
    y[idx] = v;
}

// weight (K,N) f32 -> (N,K) f16
__global__ __launch_bounds__(256)
void wtrans_kernel(const float* __restrict__ src, f16* __restrict__ dst, int K, int N) {
    const int idx = blockIdx.x * 256 + threadIdx.x;
    if (idx >= N * K) return;
    const int k = idx % K, n = idx / K;
    dst[idx] = (f16)src[(long long)k * N + n];
}

__global__ __launch_bounds__(256)
void cvt16_kernel(const float* __restrict__ src, f16* __restrict__ dst, int n) {
    const int idx = blockIdx.x * 256 + threadIdx.x;
    if (idx < n) dst[idx] = (f16)src[idx];
}

// softmax over 256 f32 scores per row -> f16 probs; one wave per row
__global__ __launch_bounds__(256)
void softmax_kernel(const float* __restrict__ S, f16* __restrict__ P, int rows) {
    const int wave = threadIdx.x >> 5, lane = threadIdx.x & 31;
    const long long r = (long long)blockIdx.x * 8 + wave;
    if (r >= rows) return;
    const float* s = S + r * TP + lane * 8;
    float4 a0 = *(const float4*)s;
    float4 a1 = *(const float4*)(s + 4);
    float v[8] = {a0.x, a0.y, a0.z, a0.w, a1.x, a1.y, a1.z, a1.w};
    float mx = v[0];
#pragma unroll
    for (int i = 1; i < 8; ++i) mx = fmaxf(mx, v[i]);
    for (int m = 16; m >= 1; m >>= 1) mx = fmaxf(mx, __shfl_xor(mx, m, 32));
    float sum = 0.f;
#pragma unroll
    for (int i = 0; i < 8; ++i) { v[i] = __expf(v[i] - mx); sum += v[i]; }
    for (int m = 16; m >= 1; m >>= 1) sum += __shfl_xor(sum, m, 32);
    const float inv = 1.0f / sum;
    union { f16 h[8]; float4 f; } o;
#pragma unroll
    for (int i = 0; i < 8; ++i) o.h[i] = (f16)(v[i] * inv);
    *(float4*)(P + r * TP + lane * 8) = o.f;
}

// v (NSEQ,TP,C) f16 -> vT (BH, DK, TP) f16  [B^T layout for P@V]
__global__ __launch_bounds__(256)
void vtrans_kernel(const f16* __restrict__ v, f16* __restrict__ vT, int n) {
    const int idx = blockIdx.x * 256 + threadIdx.x;
    if (idx >= n) return;
    const int s  = idx & 255;
    const int d  = (idx >> 8) & 63;
    const int bh = idx >> 14;
    const int bb = bh >> 2, hh = bh & 3;
    vT[idx] = v[((long long)bb * TP + s) * CDIM + hh * DK + d];
}

// GLU: (MP,512) f16 -> (MP,256) f16 : out = y[:,C:] * sigmoid(y[:,:C])
__global__ __launch_bounds__(256)
void glu_kernel(const f16* __restrict__ H, f16* __restrict__ O, int n) {
    const int idx = blockIdx.x * 256 + threadIdx.x;
    if (idx >= n) return;
    const int c = idx & 255;
    const long long row = idx >> 8;
    const float g = (float)H[row * 512 + c];
    const float u = (float)H[row * 512 + 256 + c];
    O[idx] = (f16)(u / (1.0f + __expf(-g)));
}

// depthwise conv (K=33, pad 16) + bias + BN(eval) + ReLU, f16 in/out
__global__ __launch_bounds__(256)
void dwconv_kernel(const f16* __restrict__ X, const float* __restrict__ w,
                   const float* __restrict__ wb, const float* __restrict__ bng,
                   const float* __restrict__ bnb, const float* __restrict__ bnm,
                   const float* __restrict__ bnv, f16* __restrict__ O, int n) {
    const int idx = blockIdx.x * 256 + threadIdx.x;
    if (idx >= n) return;
    const int c = idx & 255;
    const int row = idx >> 8;
    const int t = row & 255;
    const int nn = row >> 8;
    float out = 0.f;
    if (t < TREAL) {
        float acc = 0.f;
#pragma unroll
        for (int kk = 0; kk < KCONV; ++kk) {
            const int tt = t + kk - (KCONV - 1) / 2;
            if ((unsigned)tt < (unsigned)TREAL)
                acc += (float)X[((long long)nn * TP + tt) * CDIM + c] * w[c * KCONV + kk];
        }
        acc += wb[c];
        const float sc = bng[c] * rsqrtf(bnv[c] + 1e-5f);
        out = fmaxf((acc - bnm[c]) * sc + bnb[c], 0.0f);
    }
    O[idx] = (f16)out;
}

// ---------------------------------------------------------------------------
// Host orchestration
// ---------------------------------------------------------------------------
extern "C" void kernel_launch(void* const* d_in, const int* in_sizes, int n_in,
                              void* d_out, int out_size, void* d_ws, size_t ws_size,
                              hipStream_t stream) {
    // --- input pointers (dict order of setup_inputs) ---
    const float* x_in  = (const float*)d_in[0];
    const int*   maskp = (const int*)d_in[1];
    const float* fi_lng = (const float*)d_in[2],  *fi_lnb = (const float*)d_in[3];
    const float* fi_w1  = (const float*)d_in[4],  *fi_b1  = (const float*)d_in[5];
    const float* fi_w2  = (const float*)d_in[6],  *fi_b2  = (const float*)d_in[7];
    const float* at_lng = (const float*)d_in[8],  *at_lnb = (const float*)d_in[9];
    const float* wq = (const float*)d_in[10], *bq = (const float*)d_in[11];
    const float* wk = (const float*)d_in[12], *bk = (const float*)d_in[13];
    const float* wv = (const float*)d_in[14], *bv = (const float*)d_in[15];
    const float* wo = (const float*)d_in[16], *bo = (const float*)d_in[17];
    const float* cv_lng = (const float*)d_in[18], *cv_lnb = (const float*)d_in[19];
    const float* pw1w = (const float*)d_in[20], *pw1b = (const float*)d_in[21];
    const float* dww  = (const float*)d_in[22], *dwb  = (const float*)d_in[23];
    const float* bng  = (const float*)d_in[24], *bnb  = (const float*)d_in[25];
    const float* bnm  = (const float*)d_in[26], *bnv  = (const float*)d_in[27];
    const float* pw2w = (const float*)d_in[28], *pw2b = (const float*)d_in[29];
    const float* fo_lng = (const float*)d_in[30], *fo_lnb = (const float*)d_in[31];
    const float* fo_w1  = (const float*)d_in[32], *fo_b1  = (const float*)d_in[33];
    const float* fo_w2  = (const float*)d_in[34], *fo_b2  = (const float*)d_in[35];
    const float* lng = (const float*)d_in[36], *lnb = (const float*)d_in[37];
    const float* pek = (const float*)d_in[38];

    // --- workspace carve-up ---
    char* cur = (char*)d_ws;
    auto alloc = [&](size_t bytes) -> char* {
        char* p = cur; cur += (bytes + 255) & ~(size_t)255; return p;
    };
    float* yA   = (float*)alloc((size_t)MP * CDIM * 4);
    float* yB   = (float*)alloc((size_t)MP * CDIM * 4);
    f16* ln_h   = (f16*)alloc((size_t)MP * CDIM * 2);
    f16* big_h  = (f16*)alloc((size_t)MP * 1024 * 2);
    f16* q_h    = (f16*)alloc((size_t)MP * CDIM * 2);
    f16* k_h    = (f16*)alloc((size_t)MP * CDIM * 2);
    f16* v_h    = (f16*)alloc((size_t)MP * CDIM * 2);
    f16* vT_h   = (f16*)alloc((size_t)BH * DK * TP * 2);
    f16* glu_h  = (f16*)alloc((size_t)MP * CDIM * 2);
    f16* conv_h = (f16*)alloc((size_t)MP * CDIM * 2);
    f16* G_h    = (f16*)alloc((size_t)BH * TP * RP * 2);
    float* S_f  = (float*)alloc((size_t)BH * TP * TP * 4);
    f16* P_h    = (f16*)alloc((size_t)BH * TP * TP * 2);
    f16* w1iT  = (f16*)alloc((size_t)1024 * 256 * 2);
    f16* w2iT  = (f16*)alloc((size_t)256 * 1024 * 2);
    f16* wqT   = (f16*)alloc((size_t)256 * 256 * 2);
    f16* wkT   = (f16*)alloc((size_t)256 * 256 * 2);
    f16* wvT   = (f16*)alloc((size_t)256 * 256 * 2);
    f16* woT   = (f16*)alloc((size_t)256 * 256 * 2);
    f16* pw1T  = (f16*)alloc((size_t)512 * 256 * 2);
    f16* pw2T  = (f16*)alloc((size_t)256 * 256 * 2);
    f16* w1oT  = (f16*)alloc((size_t)1024 * 256 * 2);
    f16* w2oT  = (f16*)alloc((size_t)256 * 1024 * 2);
    f16* pe_h  = (f16*)alloc((size_t)4000 * DK * 2);
    f16* o_h   = q_h;   // alias: q dead once G + scores are done

    const dim3 blk(256);
    auto wt = [&](const float* s, f16* d, int K, int N) {
        wtrans_kernel<<<dim3((N * K + 255) / 256), blk, 0, stream>>>(s, d, K, N);
    };
    wt(fi_w1, w1iT, 256, 1024); wt(fi_w2, w2iT, 1024, 256);
    wt(wq, wqT, 256, 256); wt(wk, wkT, 256, 256);
    wt(wv, wvT, 256, 256); wt(wo, woT, 256, 256);
    wt(pw1w, pw1T, 256, 512); wt(pw2w, pw2T, 256, 256);
    wt(fo_w1, w1oT, 256, 1024); wt(fo_w2, w2oT, 1024, 256);
    cvt16_kernel<<<dim3((4000 * DK + 255) / 256), blk, 0, stream>>>(pek, pe_h, 4000 * DK);

    auto gemm = [&](GemmP p, int batch) {
        dim3 g(p.N / 64, p.M / 128, batch);
        gemm_wmma_kernel<<<g, blk, 0, stream>>>(p);
    };
    auto base = [&]() { GemmP p{}; p.innerB = 1; p.resScale = 1.0f; return p; };

    const int NEW = MP * CDIM;   // 16.7M element grids
    const dim3 egrid((NEW + 255) / 256);

    // ---- pad input: x -> y0 (yA) ----
    pad_input_kernel<<<egrid, blk, 0, stream>>>(x_in, yA, NEW);

    // ---- FFN_in: y1(yB) = y0 + 0.5*(relu(LN(y0)W1+b1)W2+b2) ----
    ln_kernel<<<dim3(MP / 8), blk, 0, stream>>>(yA, fi_lng, fi_lnb, ln_h, MP);
    { GemmP p = base(); p.A = ln_h; p.lda = 256; p.B = w1iT; p.ldb = 256; p.bias = fi_b1;
      p.Ch = big_h; p.ldc = 1024; p.M = MP; p.N = 1024; p.K = 256; p.epi = 1; gemm(p, 1); }
    { GemmP p = base(); p.A = big_h; p.lda = 1024; p.B = w2iT; p.ldb = 1024; p.bias = fi_b2;
      p.Cf = yB; p.R = yA; p.resScale = 0.5f; p.ldc = 256; p.M = MP; p.N = 256; p.K = 1024; gemm(p, 1); }

    // ---- Attention ----
    ln_kernel<<<dim3(MP / 8), blk, 0, stream>>>(yB, at_lng, at_lnb, ln_h, MP);
    { GemmP p = base(); p.A = ln_h; p.lda = 256; p.B = wqT; p.ldb = 256; p.bias = bq;
      p.Ch = q_h; p.ldc = 256; p.M = MP; p.N = 256; p.K = 256; gemm(p, 1); }
    { GemmP p = base(); p.A = ln_h; p.lda = 256; p.B = wkT; p.ldb = 256; p.bias = bk;
      p.Ch = k_h; p.ldc = 256; p.M = MP; p.N = 256; p.K = 256; gemm(p, 1); }
    { GemmP p = base(); p.A = ln_h; p.lda = 256; p.B = wvT; p.ldb = 256; p.bias = bv;
      p.Ch = v_h; p.ldc = 256; p.M = MP; p.N = 256; p.K = 256; gemm(p, 1); }

    // G[bh,t,r] = q[bh,t,:] . pe[1751+r,:]   (batched, shared B)
    { GemmP p = base(); p.innerB = NHEAD;
      p.A = q_h; p.lda = 256; p.sAo = (long long)TP * CDIM; p.sAi = DK;
      p.B = pe_h + 1751 * DK; p.ldb = DK;
      p.Ch = G_h; p.ldc = RP; p.sCo = (long long)NHEAD * TP * RP; p.sCi = (long long)TP * RP;
      p.M = TP; p.N = RP; p.K = DK; gemm(p, BH); }

    // scores = mask((q.k^T + G_gather) * 1/8)
    { GemmP p = base(); p.innerB = NHEAD;
      p.A = q_h; p.lda = 256; p.sAo = (long long)TP * CDIM; p.sAi = DK;
      p.B = k_h; p.ldb = 256; p.sBo = (long long)TP * CDIM; p.sBi = DK;
      p.Cf = S_f; p.ldc = TP; p.sCo = (long long)NHEAD * TP * TP; p.sCi = (long long)TP * TP;
      p.M = TP; p.N = TP; p.K = DK;
      p.epi = 2; p.G = G_h; p.sG = (long long)TP * RP; p.mask = maskp; p.scaleS = 0.125f;
      gemm(p, BH); }

    softmax_kernel<<<dim3(BH * TP / 8), blk, 0, stream>>>(S_f, P_h, BH * TP);
    vtrans_kernel<<<egrid, blk, 0, stream>>>(v_h, vT_h, BH * DK * TP);

    // O = P @ V  (written head-interleaved into o_h (NSEQ,TP,C))
    { GemmP p = base(); p.innerB = NHEAD;
      p.A = P_h; p.lda = TP; p.sAo = (long long)NHEAD * TP * TP; p.sAi = (long long)TP * TP;
      p.B = vT_h; p.ldb = TP; p.sBo = (long long)NHEAD * DK * TP; p.sBi = (long long)DK * TP;
      p.Ch = o_h; p.ldc = CDIM; p.sCo = (long long)TP * CDIM; p.sCi = DK;
      p.M = TP; p.N = DK; p.K = TP; gemm(p, BH); }

    // y2(yA) = y1 + O @ wo + bo
    { GemmP p = base(); p.A = o_h; p.lda = 256; p.B = woT; p.ldb = 256; p.bias = bo;
      p.Cf = yA; p.R = yB; p.ldc = 256; p.M = MP; p.N = 256; p.K = 256; gemm(p, 1); }

    // ---- Conv module ----
    ln_kernel<<<dim3(MP / 8), blk, 0, stream>>>(yA, cv_lng, cv_lnb, ln_h, MP);
    { GemmP p = base(); p.A = ln_h; p.lda = 256; p.B = pw1T; p.ldb = 256; p.bias = pw1b;
      p.Ch = big_h; p.ldc = 512; p.M = MP; p.N = 512; p.K = 256; gemm(p, 1); }
    glu_kernel<<<egrid, blk, 0, stream>>>(big_h, glu_h, NEW);
    dwconv_kernel<<<egrid, blk, 0, stream>>>(glu_h, dww, dwb, bng, bnb, bnm, bnv, conv_h, NEW);
    { GemmP p = base(); p.A = conv_h; p.lda = 256; p.B = pw2T; p.ldb = 256; p.bias = pw2b;
      p.Cf = yB; p.R = yA; p.ldc = 256; p.M = MP; p.N = 256; p.K = 256; gemm(p, 1); }

    // ---- FFN_out: y4(yA) = y3 + 0.5*(relu(LN(y3)W1+b1)W2+b2) ----
    ln_kernel<<<dim3(MP / 8), blk, 0, stream>>>(yB, fo_lng, fo_lnb, ln_h, MP);
    { GemmP p = base(); p.A = ln_h; p.lda = 256; p.B = w1oT; p.ldb = 256; p.bias = fo_b1;
      p.Ch = big_h; p.ldc = 1024; p.M = MP; p.N = 1024; p.K = 256; p.epi = 1; gemm(p, 1); }
    { GemmP p = base(); p.A = big_h; p.lda = 1024; p.B = w2oT; p.ldb = 1024; p.bias = fo_b2;
      p.Cf = yA; p.R = yB; p.resScale = 0.5f; p.ldc = 256; p.M = MP; p.N = 256; p.K = 1024; gemm(p, 1); }

    // ---- final LN + scatter to (B,T,C,F) ----
    ln_final_kernel<<<dim3((NSEQ * TREAL + 7) / 8), blk, 0, stream>>>(
        yA, lng, lnb, (float*)d_out, NSEQ * TREAL);
}
